// _DenseLayer_37881611550955
// MI455X (gfx1250) — matrix operand order repaired and checked by reading, not compiled, not run on hardware
//
#include <hip/hip_runtime.h>
#include <stdint.h>

// ---------- vector types for WMMA ----------
typedef __attribute__((ext_vector_type(16))) _Float16 v16h;
typedef __attribute__((ext_vector_type(8)))  _Float16 v8h;
typedef __attribute__((ext_vector_type(8)))  float    v8f;

// ---------- problem geometry ----------
constexpr int N_    = 64;
constexpr int C_IN  = 512;
constexpr int C_MID = 128;
constexpr int C_OUT = 32;
constexpr int H_    = 28;
constexpr int W_    = 28;
constexpr int HW    = H_ * W_;        // 784 (divisible by 16)
constexpr int COLS  = N_ * HW;        // 50176
constexpr int KTOT2 = C_MID * 9;      // 1152

// ---------- workspace layout (bytes) ----------
constexpr size_t OFF_MM   = 0;                                 // 4 uints: min1,max1,min2,max2
constexpr size_t OFF_INV1 = 64;                                // f32[512]
constexpr size_t OFF_ADD1 = OFF_INV1 + 512 * 4;                // f32[512]
constexpr size_t OFF_INV2 = OFF_ADD1 + 512 * 4;                // f32[128]
constexpr size_t OFF_ADD2 = OFF_INV2 + 128 * 4;                // f32[128]
constexpr size_t OFF_W1Q  = 8192;                              // f16[128*512]
constexpr size_t OFF_W2Q  = OFF_W1Q + (size_t)128 * 512 * 2;   // f16[32*1152]
constexpr size_t OFF_H1T  = 262144;                            // f16[COLS*512]  K-contiguous
constexpr size_t OFF_H2T  = OFF_H1T + (size_t)COLS * 512 * 2;  // f16[COLS*128]  channel-last

// ---------- helpers ----------
__device__ __forceinline__ float fquant(float x, float mn, float inv_s, float s) {
    return rintf((x - mn) * inv_s) * s + mn;   // RNE matches jnp.round
}

__device__ __forceinline__ v16h zero16h() {
    v16h z;
#pragma unroll
    for (int j = 0; j < 16; ++j) z[j] = (_Float16)0.f;
    return z;
}

__device__ __forceinline__ v16h load_a_frag(const _Float16* p) {
    // A fragment in this file's contiguous k order (B uses the same): [koff..koff+15]
    v8h lo = *(const v8h*)(p);
    v8h hi = *(const v8h*)(p + 8);
    v16h a;
#pragma unroll
    for (int j = 0; j < 8; ++j) { a[j] = lo[j]; a[j + 8] = hi[j]; }
    return a;
}

// Async global->LDS copy of 16B per lane (CDNA5, ASYNCcnt-tracked).
// lds_off: wave-relative LDS byte offset = low 32 bits of flat &shared (ISA 10.2).
__device__ __forceinline__ void async_copy_b128(unsigned lds_off, const void* gsrc) {
    asm volatile("global_load_async_to_lds_b128 %0, %1, off"
                 :: "v"(lds_off), "v"(gsrc) : "memory");
}
__device__ __forceinline__ void wait_async_le4() {
    asm volatile("s_wait_asynccnt 0x4" ::: "memory");
}
__device__ __forceinline__ void wait_async_0() {
    asm volatile("s_wait_asynccnt 0x0" ::: "memory");
}

// ---------- K0: fold BN params, init min/max ----------
__global__ void k0_init(const float* __restrict__ g1, const float* __restrict__ b1,
                        const float* __restrict__ m1, const float* __restrict__ v1,
                        const float* __restrict__ g2, const float* __restrict__ b2,
                        const float* __restrict__ m2, const float* __restrict__ v2,
                        unsigned* __restrict__ mm,
                        float* __restrict__ inv1, float* __restrict__ add1,
                        float* __restrict__ inv2, float* __restrict__ add2) {
    int t = threadIdx.x;
    if (t < 4) mm[t] = (t == 0 || t == 2) ? 0x7F800000u : 0u;  // min=+inf, max=0
    if (t < 512) {
        float iv = g1[t] / sqrtf(v1[t] + 1e-5f);
        inv1[t] = iv;
        add1[t] = b1[t] - m1[t] * iv;
    }
    if (t < 128) {
        float iv = g2[t] / sqrtf(v2[t] + 1e-5f);
        inv2[t] = iv;
        add2[t] = b2[t] - m2[t] * iv;
    }
}

// ---------- K1: h1 = relu(bn1(x)) -> f16 K-contiguous + min/max ----------
__global__ void k1_bn_relu_pack(const float* __restrict__ x,
                                const float* __restrict__ inv1,
                                const float* __restrict__ add1,
                                _Float16* __restrict__ h1t,
                                unsigned* __restrict__ mm) {
    const long long total = (long long)N_ * C_IN * HW;
    float lmin = __builtin_inff(), lmax = 0.f;
    long long stride = (long long)gridDim.x * blockDim.x;
    for (long long i = (long long)blockIdx.x * blockDim.x + threadIdx.x; i < total; i += stride) {
        int hw = (int)(i % HW);
        int c  = (int)((i / HW) % C_IN);
        int n  = (int)(i / ((long long)HW * C_IN));
        float v = fmaxf(x[i] * inv1[c] + add1[c], 0.f);
        lmin = fminf(lmin, v);
        lmax = fmaxf(lmax, v);
        h1t[(long long)(n * HW + hw) * C_IN + c] = (_Float16)v;
    }
#pragma unroll
    for (int m = 16; m > 0; m >>= 1) {
        lmin = fminf(lmin, __shfl_xor(lmin, m, 32));
        lmax = fmaxf(lmax, __shfl_xor(lmax, m, 32));
    }
    if ((threadIdx.x & 31) == 0) {
        atomicMin(&mm[0], __float_as_uint(lmin));   // valid: all values >= 0
        atomicMax(&mm[1], __float_as_uint(lmax));
    }
}

// ---------- K2/K4: weight min/max + quantize to f16 (mode 1 reorders w2) ----------
__global__ void k_wquant(const float* __restrict__ w, _Float16* __restrict__ wq,
                         int nelem, int mode) {
    __shared__ float smin[32], smax[32];
    float lmin = __builtin_inff(), lmax = -__builtin_inff();
    for (int i = threadIdx.x; i < nelem; i += blockDim.x) {
        float v = w[i];
        lmin = fminf(lmin, v);
        lmax = fmaxf(lmax, v);
    }
#pragma unroll
    for (int m = 16; m > 0; m >>= 1) {
        lmin = fminf(lmin, __shfl_xor(lmin, m, 32));
        lmax = fmaxf(lmax, __shfl_xor(lmax, m, 32));
    }
    int wid = threadIdx.x >> 5, lane = threadIdx.x & 31;
    if (lane == 0) { smin[wid] = lmin; smax[wid] = lmax; }
    __syncthreads();
    if (wid == 0) {
        int nw = (blockDim.x + 31) >> 5;
        lmin = (lane < nw) ? smin[lane] : __builtin_inff();
        lmax = (lane < nw) ? smax[lane] : -__builtin_inff();
#pragma unroll
        for (int m = 16; m > 0; m >>= 1) {
            lmin = fminf(lmin, __shfl_xor(lmin, m, 32));
            lmax = fmaxf(lmax, __shfl_xor(lmax, m, 32));
        }
        if (lane == 0) { smin[0] = lmin; smax[0] = lmax; }
    }
    __syncthreads();
    float mn = smin[0], mx = smax[0];
    float s = fmaxf((mx - mn) * (1.f / 255.f), 1e-8f);
    float is = 1.f / s;
    for (int i = threadIdx.x; i < nelem; i += blockDim.x) {
        float q = fquant(w[i], mn, is, s);
        int o = i;
        if (mode) {  // w2 OIHW [32][128][3][3] -> [m][(dy*3+dx)*128 + c]
            int mo = i / KTOT2;
            int r  = i % KTOT2;
            int c  = r / 9;
            int dy = (r % 9) / 3;
            int dx = r % 3;
            o = mo * KTOT2 + (dy * 3 + dx) * C_MID + c;
        }
        wq[o] = (_Float16)q;
    }
}

// ---------- K3: GEMM1 (1x1 conv 512->128) via WMMA ----------
// Wave = 2 row-tiles x 4 col-tiles. A slab staged via async->LDS (double buffered,
// per-wave -> no barriers); B double-buffered in registers.
__global__ void __launch_bounds__(256)
k3_gemm1(const _Float16* __restrict__ h1t, const _Float16* __restrict__ w1q,
         const float* __restrict__ inv2, const float* __restrict__ add2,
         _Float16* __restrict__ h2t, unsigned* __restrict__ mm) {
    __shared__ alignas(16) _Float16 aslab[8][2][32 * 32];   // 8 waves x 2 bufs x 2KB

    int lane = threadIdx.x & 31;
    int wid  = threadIdx.x >> 5;
    int rp   = wid & 3;                 // rows  32*rp .. 32*rp+31
    int cq   = wid >> 2;                // cols  blockIdx*128 + cq*64 .. +63
    int rowbase = rp * 32;
    int colbase = blockIdx.x * 128 + cq * 64;
    int lr = lane & 15;
    int hi = lane >> 4;

    float mn = __uint_as_float(mm[0]);
    float mx = __uint_as_float(mm[1]);
    float s  = fmaxf((mx - mn) * (1.f / 255.f), 1e-8f);
    float is = 1.f / s;

    // A staging: lane copies row (rowbase+lane), 4x16B chunks per buffer
    const _Float16* asrc = w1q + (rowbase + lane) * C_IN;
    unsigned adst0 = (unsigned)(uintptr_t)&aslab[wid][0][lane * 32];
    unsigned adst1 = (unsigned)(uintptr_t)&aslab[wid][1][lane * 32];

    // B pointers: 4 column tiles, K-contiguous
    const _Float16* bptr[4];
#pragma unroll
    for (int t = 0; t < 4; ++t)
        bptr[t] = h1t + (long long)(colbase + t * 16 + lr) * C_IN + hi * 16;

    v8f acc[2][4];
#pragma unroll
    for (int rt = 0; rt < 2; ++rt)
#pragma unroll
        for (int t = 0; t < 4; ++t)
#pragma unroll
            for (int j = 0; j < 8; ++j) acc[rt][t][j] = 0.f;

    // prologue: stage A buf0 (k0=0), load raw B k0=0
#pragma unroll
    for (int c = 0; c < 4; ++c)
        async_copy_b128(adst0 + c * 16, asrc + 0 + c * 8);
    v16h Braw[4];
#pragma unroll
    for (int t = 0; t < 4; ++t) Braw[t] = *(const v16h*)(bptr[t]);

    for (int k0 = 0; k0 < C_IN; k0 += 32) {
        int buf = (k0 >> 5) & 1;
        bool hasnext = (k0 + 32) < C_IN;

        // issue next-stage async A copies + next raw B loads
        v16h Bn[4];
        if (hasnext) {
            unsigned adst = buf ? adst0 : adst1;
#pragma unroll
            for (int c = 0; c < 4; ++c)
                async_copy_b128(adst + c * 16, asrc + (k0 + 32) + c * 8);
#pragma unroll
            for (int t = 0; t < 4; ++t)
                Bn[t] = *(const v16h*)(bptr[t] + k0 + 32);
        }

        // quantize current B in place (VALU hides async/load latency)
#pragma unroll
        for (int t = 0; t < 4; ++t)
#pragma unroll
            for (int j = 0; j < 16; ++j)
                Braw[t][j] = (_Float16)fquant((float)Braw[t][j], mn, is, s);

        // retire current A buffer (newest 4 copies may stay in flight)
        if (hasnext) wait_async_le4(); else wait_async_0();

        const _Float16* sa = &aslab[wid][buf][0];
        v16h A0 = load_a_frag(sa + lr * 32 + hi * 16);
        v16h A1 = load_a_frag(sa + (16 + lr) * 32 + hi * 16);

#pragma unroll
        for (int t = 0; t < 4; ++t) {
            acc[0][t] = __builtin_amdgcn_wmma_f32_16x16x32_f16(
                false, A0, false, Braw[t], (short)0, acc[0][t], false, false);
            acc[1][t] = __builtin_amdgcn_wmma_f32_16x16x32_f16(
                false, A1, false, Braw[t], (short)0, acc[1][t], false, false);
        }

        if (hasnext) {
#pragma unroll
            for (int t = 0; t < 4; ++t) Braw[t] = Bn[t];
        }
    }

    // fused bn2 + relu, f16 channel-last store, min/max reduce
    float lmin = __builtin_inff(), lmax = 0.f;
    int mb = hi * 8;
#pragma unroll
    for (int t = 0; t < 4; ++t) {
        long long col = colbase + t * 16 + lr;
#pragma unroll
        for (int rt = 0; rt < 2; ++rt) {
            v8h o;
#pragma unroll
            for (int v = 0; v < 8; ++v) {
                int m = rowbase + rt * 16 + mb + v;
                float y = fmaxf(acc[rt][t][v] * inv2[m] + add2[m], 0.f);
                lmin = fminf(lmin, y);
                lmax = fmaxf(lmax, y);
                o[v] = (_Float16)y;
            }
            *(v8h*)(h2t + col * C_MID + rowbase + rt * 16 + mb) = o;
        }
    }
#pragma unroll
    for (int m = 16; m > 0; m >>= 1) {
        lmin = fminf(lmin, __shfl_xor(lmin, m, 32));
        lmax = fmaxf(lmax, __shfl_xor(lmax, m, 32));
    }
    if (lane == 0) {
        atomicMin(&mm[2], __float_as_uint(lmin));
        atomicMax(&mm[3], __float_as_uint(lmax));
    }
}

// ---------- K5: implicit 3x3 conv 128->32 via WMMA, 1-step B look-ahead ----------
__global__ void __launch_bounds__(256)
k5_conv2(const _Float16* __restrict__ h2t, const _Float16* __restrict__ w2q,
         float* __restrict__ out, const unsigned* __restrict__ mm) {
    int lane = threadIdx.x & 31;
    int wid  = threadIdx.x >> 5;
    int tile = blockIdx.x * 8 + wid;
    int col0 = tile * 16;
    int lr = lane & 15;
    int hi = lane >> 4;

    float mn = __uint_as_float(mm[2]);
    float mx = __uint_as_float(mm[3]);
    float s  = fmaxf((mx - mn) * (1.f / 255.f), 1e-8f);
    float is = 1.f / s;

    int col = col0 + lr;
    int n   = col / HW;
    int hw  = col % HW;
    int hy  = hw / W_, hx = hw % W_;

    const _Float16* bb[9];
    bool vld[9];
#pragma unroll
    for (int e = 0; e < 9; ++e) {
        int dy = e / 3, dx = e % 3;
        int yy = hy + dy - 1, xx = hx + dx - 1;
        vld[e] = (yy >= 0) & (yy < H_) & (xx >= 0) & (xx < W_);
        bb[e]  = h2t + (long long)(n * HW + yy * W_ + xx) * C_MID + hi * 16;
    }
    const _Float16* abase0 = w2q + lr * KTOT2 + hi * 16;
    const _Float16* abase1 = w2q + (16 + lr) * KTOT2 + hi * 16;

    v8f acc0, acc1;
#pragma unroll
    for (int j = 0; j < 8; ++j) { acc0[j] = 0.f; acc1[j] = 0.f; }

    v16h Braw = vld[0] ? *(const v16h*)(bb[0]) : zero16h();
    bool cv = vld[0];

#pragma unroll
    for (int sstep = 0; sstep < 36; ++sstep) {
        int e  = sstep >> 2;
        int k0 = (sstep & 3) * 32;
        int kb = e * C_MID + k0;

        // look-ahead raw B for step s+1
        v16h Bn = zero16h();
        bool nv = false;
        if (sstep + 1 < 36) {
            int e2 = (sstep + 1) >> 2;
            int k2 = ((sstep + 1) & 3) * 32;
            nv = vld[e2];
            if (nv) Bn = *(const v16h*)(bb[e2] + k2);
        }
        __builtin_prefetch(abase0 + kb + 64, 0, 1);

        v16h a0 = load_a_frag(abase0 + kb);
        v16h a1 = load_a_frag(abase1 + kb);

        if (cv) {   // padding lanes stay exact zero (ref pads after quantization)
#pragma unroll
            for (int j = 0; j < 16; ++j)
                Braw[j] = (_Float16)fquant((float)Braw[j], mn, is, s);
        }

        acc0 = __builtin_amdgcn_wmma_f32_16x16x32_f16(
            false, a0, false, Braw, (short)0, acc0, false, false);
        acc1 = __builtin_amdgcn_wmma_f32_16x16x32_f16(
            false, a1, false, Braw, (short)0, acc1, false, false);

        Braw = Bn;
        cv = nv;
    }

    int mb = hi * 8;
#pragma unroll
    for (int v = 0; v < 8; ++v) {
        out[(long long)(n * C_OUT + mb + v) * HW + hw]      = acc0[v];
        out[(long long)(n * C_OUT + 16 + mb + v) * HW + hw] = acc1[v];
    }
}

// ---------- launcher ----------
extern "C" void kernel_launch(void* const* d_in, const int* in_sizes, int n_in,
                              void* d_out, int out_size, void* d_ws, size_t ws_size,
                              hipStream_t stream) {
    (void)in_sizes; (void)n_in; (void)out_size; (void)ws_size;
    const float* x  = (const float*)d_in[0];
    const float* g1 = (const float*)d_in[1];
    const float* b1 = (const float*)d_in[2];
    const float* m1 = (const float*)d_in[3];
    const float* v1 = (const float*)d_in[4];
    const float* w1 = (const float*)d_in[5];
    const float* g2 = (const float*)d_in[6];
    const float* b2 = (const float*)d_in[7];
    const float* m2 = (const float*)d_in[8];
    const float* v2 = (const float*)d_in[9];
    const float* w2 = (const float*)d_in[10];

    char* ws = (char*)d_ws;
    unsigned* mm  = (unsigned*)(ws + OFF_MM);
    float* inv1   = (float*)(ws + OFF_INV1);
    float* add1   = (float*)(ws + OFF_ADD1);
    float* inv2   = (float*)(ws + OFF_INV2);
    float* add2   = (float*)(ws + OFF_ADD2);
    _Float16* w1q = (_Float16*)(ws + OFF_W1Q);
    _Float16* w2q = (_Float16*)(ws + OFF_W2Q);
    _Float16* h1t = (_Float16*)(ws + OFF_H1T);
    _Float16* h2t = (_Float16*)(ws + OFF_H2T);

    k0_init<<<1, 512, 0, stream>>>(g1, b1, m1, v1, g2, b2, m2, v2,
                                   mm, inv1, add1, inv2, add2);
    k1_bn_relu_pack<<<4096, 256, 0, stream>>>(x, inv1, add1, h1t, mm);
    k_wquant<<<1, 1024, 0, stream>>>(w1, w1q, 128 * 512, 0);
    k3_gemm1<<<COLS / 128, 256, 0, stream>>>(h1t, w1q, inv2, add2, h2t, mm);
    k_wquant<<<1, 1024, 0, stream>>>(w2, w2q, 32 * KTOT2, 1);
    k5_conv2<<<COLS / 16 / 8, 256, 0, stream>>>(h2t, w2q, (float*)d_out, mm);
}
